// SequentialGaussianBNN_3307124817925
// MI455X (gfx1250) — matrix operand order repaired and checked by reading, not compiled, not run on hardware
//
#include <hip/hip_runtime.h>

// ---- types ---------------------------------------------------------------
typedef __bf16 bf16_t;
typedef __attribute__((ext_vector_type(4)))  bf16_t bf16x4;
typedef __attribute__((ext_vector_type(8)))  bf16_t bf16x8;
typedef __attribute__((ext_vector_type(16))) bf16_t v16bf;
typedef __attribute__((ext_vector_type(8)))  float  v8f;

__device__ __forceinline__ float splus(float x) {
    // numerically stable softplus: max(x,0) + log1p(exp(-|x|))
    return fmaxf(x, 0.0f) + log1pf(__expf(-fabsf(x)));
}

// ---- weight materialization: A = bf16(mW + softplus(sW)*zW) --------------
__global__ __launch_bounds__(256) void prep_weight_kernel(
    const float* __restrict__ mW, const float* __restrict__ sW,
    const float* __restrict__ zW, bf16_t* __restrict__ A, int n4)
{
    int i = blockIdx.x * blockDim.x + threadIdx.x;
    if (i >= n4) return;
    float4 m = ((const float4*)mW)[i];
    float4 s = ((const float4*)sW)[i];
    float4 z = ((const float4*)zW)[i];
    bf16x4 o;
    o[0] = (bf16_t)(m.x + splus(s.x) * z.x);
    o[1] = (bf16_t)(m.y + splus(s.y) * z.y);
    o[2] = (bf16_t)(m.z + splus(s.z) * z.z);
    o[3] = (bf16_t)(m.w + splus(s.w) * z.w);
    *(bf16x4*)&A[(size_t)i * 4] = o;
}

// ---- fp32 -> bf16 convert (for input x) ----------------------------------
__global__ __launch_bounds__(256) void cvt_bf16_kernel(
    const float* __restrict__ in, bf16_t* __restrict__ out, int n4)
{
    int i = blockIdx.x * blockDim.x + threadIdx.x;
    if (i >= n4) return;
    float4 v = ((const float4*)in)[i];
    bf16x4 o;
    o[0] = (bf16_t)v.x; o[1] = (bf16_t)v.y; o[2] = (bf16_t)v.z; o[3] = (bf16_t)v.w;
    *(bf16x4*)&out[(size_t)i * 4] = o;
}

// ---- WMMA GEMM: out[M][N] = act( X[M][K] * W[N][K]^T + bias(N) ) ---------
// BM=128, BN=128, BK=32. 256 threads = 8 waves (2 along M x 4 along N).
// Each wave owns a 64x32 tile = 4x2 v_wmma_f32_16x16x32_bf16 (8 acc of v8f).
// LDS is double-buffered and filled with GLOBAL_LOAD_ASYNC_TO_LDS_B128
// (ASYNCcnt-tracked); s_wait_asynccnt 0 precedes the single per-iteration
// barrier, and the next buffer's async fills overlap the WMMA phase.
template <bool RELU, bool OUTBF16>
__global__ __launch_bounds__(256) void bnn_gemm_kernel(
    const bf16_t* __restrict__ X, const bf16_t* __restrict__ W,
    const float* __restrict__ mb, const float* __restrict__ sb,
    const float* __restrict__ zb, void* __restrict__ outp,
    int M, int N, int K)
{
    constexpr int BM = 128, BN = 128, BK = 32;
    constexpr int BUFB = BM * BK * 2;                  // 8192 B per buffer
    __shared__ __align__(16) bf16_t Xs[2][BM * BK];    // 2 x 8 KB
    __shared__ __align__(16) bf16_t Ws[2][BN * BK];    // 2 x 8 KB

    const int t     = threadIdx.x;
    const int lane  = t & 31;
    const int wave  = t >> 5;
    const int mbase = (wave & 1) * 64;     // 2 waves along M, 64 rows each
    const int nbase = (wave >> 1) * 32;    // 4 waves along N, 32 cols each
    const int m0    = blockIdx.y * BM;
    const int n0    = blockIdx.x * BN;
    const int lhalf = lane >> 4;           // lane half (0|1)
    const int l15   = lane & 15;

    // staging: 128x32 bf16 tile = 8KB = 32B/thread -> row = t/2, one 32B span
    const int sr = t >> 1, sc = (t & 1) * 16;

    const bf16_t* gX = X + (size_t)(m0 + sr) * K + sc;
    const bf16_t* gW = W + (size_t)(n0 + sr) * K + sc;

    // 32-bit LDS byte addresses of this thread's staging span (buffer 0)
    const unsigned ldsX0 = (unsigned)(uintptr_t)&Xs[0][sr * BK + sc];
    const unsigned ldsW0 = (unsigned)(uintptr_t)&Ws[0][sr * BK + sc];

    v8f acc[4][2] = {};

    // async-fill of K-tile 0 into buffer 0 (2x b128 per tile per thread;
    // the inst offset applies to BOTH the LDS and the global address)
    asm volatile(
        "global_load_async_to_lds_b128 %0, %2, off\n\t"
        "global_load_async_to_lds_b128 %0, %2, off offset:16\n\t"
        "global_load_async_to_lds_b128 %1, %3, off\n\t"
        "global_load_async_to_lds_b128 %1, %3, off offset:16"
        :: "v"(ldsX0), "v"(ldsW0), "v"(gX), "v"(gW) : "memory");

    const int nk = K / BK;
    for (int kt = 0; kt < nk; ++kt) {
        const int cur = kt & 1;

        asm volatile("s_wait_asynccnt 0x0" ::: "memory");  // buffer `cur` filled
        __syncthreads();                                   // ... and published

        // kick off async fill of the other buffer (everyone is done reading it)
        if (kt + 1 < nk) {
            gX += BK; gW += BK;
            const unsigned lx = ldsX0 + (unsigned)((cur ^ 1) * BUFB);
            const unsigned lw = ldsW0 + (unsigned)((cur ^ 1) * BUFB);
            asm volatile(
                "global_load_async_to_lds_b128 %0, %2, off\n\t"
                "global_load_async_to_lds_b128 %0, %2, off offset:16\n\t"
                "global_load_async_to_lds_b128 %1, %3, off\n\t"
                "global_load_async_to_lds_b128 %1, %3, off offset:16"
                :: "v"(lx), "v"(lw), "v"(gX), "v"(gW) : "memory");
            if (kt + 2 < nk) {
                __builtin_prefetch(gX + BK, 0, 3);   // global_prefetch_b8
                __builtin_prefetch(gW + BK, 0, 3);
            }
        }

        // load ALL fragments (one DS-read group), then 8 back-to-back WMMAs
        v16bf a[4], b[2];
#pragma unroll
        for (int nt = 0; nt < 2; ++nt) {
            int row = nbase + nt * 16 + l15;
            bf16x8 lo = *(const bf16x8*)&Ws[cur][row * BK + lhalf * 16];
            bf16x8 hi = *(const bf16x8*)&Ws[cur][row * BK + lhalf * 16 + 8];
            b[nt] = __builtin_shufflevector(lo, hi, 0,1,2,3,4,5,6,7,8,9,10,11,12,13,14,15);
        }
#pragma unroll
        for (int mt = 0; mt < 4; ++mt) {
            int row = mbase + mt * 16 + l15;
            bf16x8 lo = *(const bf16x8*)&Xs[cur][row * BK + lhalf * 16];
            bf16x8 hi = *(const bf16x8*)&Xs[cur][row * BK + lhalf * 16 + 8];
            a[mt] = __builtin_shufflevector(lo, hi, 0,1,2,3,4,5,6,7,8,9,10,11,12,13,14,15);
        }
#pragma unroll
        for (int mt = 0; mt < 4; ++mt) {
            acc[mt][0] = __builtin_amdgcn_wmma_f32_16x16x32_bf16(
                false, a[mt], false, b[0], (short)0, acc[mt][0], false, false);
            acc[mt][1] = __builtin_amdgcn_wmma_f32_16x16x32_bf16(
                false, a[mt], false, b[1], (short)0, acc[mt][1], false, false);
        }

        // scheduler guidance: cluster the 12 DS reads, then the 8 WMMAs
        __builtin_amdgcn_sched_group_barrier(0x100, 12, 0);  // DS read
        __builtin_amdgcn_sched_group_barrier(0x008,  8, 0);  // MFMA/WMMA
    }

    // epilogue: bias = mb + softplus(sb)*zb, optional ReLU
    float bias[2];
#pragma unroll
    for (int nt = 0; nt < 2; ++nt) {
        int col = n0 + nbase + nt * 16 + l15;
        bias[nt] = mb[col] + splus(sb[col]) * zb[col];
    }
#pragma unroll
    for (int mt = 0; mt < 4; ++mt)
#pragma unroll
        for (int nt = 0; nt < 2; ++nt) {
            int col = n0 + nbase + nt * 16 + l15;
#pragma unroll
            for (int r = 0; r < 8; ++r) {
                // C/D layout: VGPR r, lanes 0-15 -> M=r, lanes 16-31 -> M=r+8
                int row = m0 + mbase + mt * 16 + lhalf * 8 + r;
                float v = acc[mt][nt][r] + bias[nt];
                if (RELU) v = fmaxf(v, 0.0f);
                if (OUTBF16) ((bf16_t*)outp)[(size_t)row * N + col] = (bf16_t)v;
                else         ((float*)outp)[(size_t)row * N + col]  = v;
            }
        }
}

// ---- host launch ---------------------------------------------------------
extern "C" void kernel_launch(void* const* d_in, const int* in_sizes, int n_in,
                              void* d_out, int out_size, void* d_ws, size_t ws_size,
                              hipStream_t stream)
{
    constexpr int BATCH = 8192;
    constexpr int D0 = 1024, D1 = 4096, D2 = 4096, D3 = 1024;

    const float* x = (const float*)d_in[0];
    const float* mW[3] = { (const float*)d_in[1],  (const float*)d_in[7],  (const float*)d_in[13] };
    const float* sW[3] = { (const float*)d_in[2],  (const float*)d_in[8],  (const float*)d_in[14] };
    const float* zW[3] = { (const float*)d_in[3],  (const float*)d_in[9],  (const float*)d_in[15] };
    const float* mb[3] = { (const float*)d_in[4],  (const float*)d_in[10], (const float*)d_in[16] };
    const float* sb[3] = { (const float*)d_in[5],  (const float*)d_in[11], (const float*)d_in[17] };
    const float* zb[3] = { (const float*)d_in[6],  (const float*)d_in[12], (const float*)d_in[18] };

    // workspace layout (bf16), all 16B-aligned
    bf16_t* A0 = (bf16_t*)d_ws;
    bf16_t* A1 = A0 + (size_t)D1 * D0;       // 4096*1024
    bf16_t* A2 = A1 + (size_t)D2 * D1;       // 4096*4096
    bf16_t* Xb = A2 + (size_t)D3 * D2;       // 1024*4096
    bf16_t* H0 = Xb + (size_t)BATCH * D0;    // 8192*1024
    bf16_t* H1 = H0 + (size_t)BATCH * D1;    // 8192*4096

    // 1) materialize bf16 weights
    {
        int n4 = (D1 * D0) / 4;
        prep_weight_kernel<<<n4 / 256, 256, 0, stream>>>(mW[0], sW[0], zW[0], A0, n4);
    }
    {
        int n4 = (D2 * D1) / 4;
        prep_weight_kernel<<<n4 / 256, 256, 0, stream>>>(mW[1], sW[1], zW[1], A1, n4);
    }
    {
        int n4 = (D3 * D2) / 4;
        prep_weight_kernel<<<n4 / 256, 256, 0, stream>>>(mW[2], sW[2], zW[2], A2, n4);
    }
    // 2) convert x to bf16
    {
        int n4 = (BATCH * D0) / 4;
        cvt_bf16_kernel<<<n4 / 256, 256, 0, stream>>>(x, Xb, n4);
    }
    // 3) layer 0: [8192,1024] x [4096,1024]^T -> relu -> H0 bf16
    bnn_gemm_kernel<true, true><<<dim3(D1 / 128, BATCH / 128), 256, 0, stream>>>(
        Xb, A0, mb[0], sb[0], zb[0], H0, BATCH, D1, D0);
    // 4) layer 1: [8192,4096] x [4096,4096]^T -> relu -> H1 bf16
    bnn_gemm_kernel<true, true><<<dim3(D2 / 128, BATCH / 128), 256, 0, stream>>>(
        H0, A1, mb[1], sb[1], zb[1], H1, BATCH, D2, D1);
    // 5) layer 2: [8192,4096] x [1024,4096]^T -> fp32 d_out
    bnn_gemm_kernel<false, false><<<dim3(D3 / 128, BATCH / 128), 256, 0, stream>>>(
        H1, A2, mb[2], sb[2], zb[2], d_out, BATCH, D3, D2);
}
